// Codebook_12180527251874
// MI455X (gfx1250) — matrix operand order repaired and checked by reading, not compiled, not run on hardware
//
#include <hip/hip_runtime.h>
#include <stdint.h>

// Problem constants (from reference): z (16,256,32,32) f32, embedding (8192,256) f32
#define KCB   8192          // codebook entries
#define DIM   256           // latent dim
#define NROW  16384         // 16*32*32 flattened z vectors
#define HW    1024          // 32*32
#define BETA  0.25f

typedef __attribute__((ext_vector_type(16))) __bf16 v16bf;
typedef __attribute__((ext_vector_type(8)))  float  v8f;

__device__ __forceinline__ uint16_t f32_to_bf16_rne(float f) {
  union { float f; uint32_t u; } v; v.f = f;
  uint32_t u = v.u;
  uint32_t r = u + 0x7FFFu + ((u >> 16) & 1u);   // round-to-nearest-even
  return (uint16_t)(r >> 16);
}
__device__ __forceinline__ float bf16_bits_to_f32(uint16_t h) {
  union { uint32_t u; float f; } v; v.u = ((uint32_t)h) << 16;
  return v.f;
}
// WGP-scope prefetch (scope field 0 = WGP -> pulls into all cache levels,
// including WGP$). __builtin_prefetch can only reach SE/SYS scopes.
__device__ __forceinline__ void prefetch_wgp(const void* p) {
  asm volatile("global_prefetch_b8 %0, off" :: "v"(p));
}

// ---------------------------------------------------------------------------
// Kernel 1: split embedding into bf16 hi/lo and compute ||e_k||^2 (fp32).
// grid = KCB blocks, 256 threads (one thread per dim element).
// ---------------------------------------------------------------------------
__global__ void prep_embedding_kernel(const float* __restrict__ emb,
                                      uint16_t* __restrict__ eh,
                                      uint16_t* __restrict__ el,
                                      float* __restrict__ enorm) {
  const int k = blockIdx.x;
  const int d = threadIdx.x;
  const float f = emb[(size_t)k * DIM + d];
  const uint16_t hb = f32_to_bf16_rne(f);
  const float    hf = bf16_bits_to_f32(hb);
  const uint16_t lb = f32_to_bf16_rne(f - hf);   // exact residual, then rne
  eh[(size_t)k * DIM + d] = hb;
  el[(size_t)k * DIM + d] = lb;

  __shared__ float sm[DIM];
  sm[d] = f * f;
  __syncthreads();
  for (int s = DIM / 2; s > 0; s >>= 1) {
    if (d < s) sm[d] += sm[d + s];
    __syncthreads();
  }
  if (d == 0) enorm[k] = sm[0];
}

// ---------------------------------------------------------------------------
// Kernel 2: transpose z (b,d,h,w) -> row-major (N, D) and split into bf16 hi/lo.
// grid = NROW blocks, 256 threads.
// ---------------------------------------------------------------------------
__global__ void prep_z_kernel(const float* __restrict__ z,
                              uint16_t* __restrict__ zh,
                              uint16_t* __restrict__ zl) {
  const int n  = blockIdx.x;       // 0..NROW-1 ; n = b*HW + hw
  const int d  = threadIdx.x;      // 0..255
  const int b  = n >> 10;
  const int hw = n & 1023;
  const float f = z[((size_t)(b * DIM + d)) * HW + hw];
  const uint16_t hb = f32_to_bf16_rne(f);
  const float    hf = bf16_bits_to_f32(hb);
  const uint16_t lb = f32_to_bf16_rne(f - hf);
  zh[(size_t)n * DIM + d] = hb;
  zl[(size_t)n * DIM + d] = lb;
}

// ---------------------------------------------------------------------------
// Kernel 3: main WMMA distance + argmin kernel.
// Grid = 256 blocks (fills ~256 WGPs), block = 256 threads = 8 waves.
// Each block owns 4 M-tiles (64 z rows). Waves 0-3 cover M-tiles m0..m3 over
// codebook half 0 (entries 0..4095); waves 4-7 cover the same M-tiles over
// half 1 (4096..8191). The two halves' per-row argmin candidates are merged
// through LDS at the end (half 0 wins ties -> first-hit argmin semantics).
// A (z tile, hi+lo) stays register-resident; split-bf16 GEMM:
//   acc += Ah*Bh + Ah*Bl + Al*Bh   (fp32 accumulate, ~17 mantissa bits)
// score = ||e||^2 - 2*dot  (row-constant ||z||^2 dropped — argmin invariant).
// B tiles are shared by waves through the WGP$; the next contiguous 8 KB tile
// is prefetched WGP-scope with full line coverage (256 B per lane).
// ---------------------------------------------------------------------------
__global__ void __launch_bounds__(256)
vq_argmin_kernel(const uint16_t* __restrict__ zh,
                 const uint16_t* __restrict__ zl,
                 const uint16_t* __restrict__ eh,
                 const uint16_t* __restrict__ el,
                 const float* __restrict__ enorm,
                 int* __restrict__ best_idx) {
  const int lane    = threadIdx.x & 31;
  const int wave    = threadIdx.x >> 5;      // 0..7
  const int mslot   = wave & 3;              // which of the block's 4 M-tiles
  const int half    = wave >> 2;             // which codebook half
  const int mtile   = blockIdx.x * 4 + mslot;
  const int rowBase = mtile * 16;

  // ISA 16-bit A layout: lanes 0-15 hold K=0..7 & 16..23, lanes 16-31 hold
  // K=8..15 & 24..31 of row (lane&15).
  const int arow = rowBase + (lane & 15);
  const int aoff = (lane >> 4) * 8;
  // ISA 16-bit B layout: lanes 0-15 hold K=0..7 & 16..23, lanes 16-31 hold
  // K=8..15 & 24..31 of column (lane&15).
  const int bcol = lane & 15;
  const int boff = (lane >> 4) * 8;

  // Load A fragments (hi+lo) for all 8 K-chunks into registers (128 VGPRs).
  v16bf a_hi[8], a_lo[8];
#pragma unroll
  for (int c = 0; c < 8; ++c) {
    const uint16_t* ph = zh + (size_t)arow * DIM + c * 32 + aoff;
    const uint16_t* pl = zl + (size_t)arow * DIM + c * 32 + aoff;
    uint4* dh = reinterpret_cast<uint4*>(&a_hi[c]);
    dh[0] = *reinterpret_cast<const uint4*>(ph);
    dh[1] = *reinterpret_cast<const uint4*>(ph + 16);
    uint4* dl = reinterpret_cast<uint4*>(&a_lo[c]);
    dl[0] = *reinterpret_cast<const uint4*>(pl);
    dl[1] = *reinterpret_cast<const uint4*>(pl + 16);
  }

  float minv[8];
  int   mini[8];
#pragma unroll
  for (int j = 0; j < 8; ++j) { minv[j] = 3.0e38f; mini[j] = 0; }

  // Per-lane slice of the next 8 KB B tile for full-coverage prefetch:
  // lane covers 256 B (= 128 uint16), two prefetches 128 B apart.
  const int poff   = lane * 128;
  const int ntBase = half * (KCB / 32);      // 256 N-tiles per half

  for (int nt = 0; nt < KCB / 32; ++nt) {
    const int nBase = (ntBase + nt) * 16;
    const uint16_t* bh_row = eh + (size_t)(nBase + bcol) * DIM + boff;
    const uint16_t* bl_row = el + (size_t)(nBase + bcol) * DIM + boff;

    // Prefetch the entire next 16x256 bf16 tile (contiguous 8 KB per array)
    // into the WGP cache, shared by the block's waves.
    const uint16_t* nh = eh + (size_t)(nBase + 16) * DIM;
    const uint16_t* nl = el + (size_t)(nBase + 16) * DIM;
    prefetch_wgp(nh + poff);
    prefetch_wgp(nh + poff + 64);
    prefetch_wgp(nl + poff);
    prefetch_wgp(nl + poff + 64);

    v8f acc = {0.f, 0.f, 0.f, 0.f, 0.f, 0.f, 0.f, 0.f};
#pragma unroll
    for (int c = 0; c < 8; ++c) {
      v16bf bh, bl;
      uint4* pb = reinterpret_cast<uint4*>(&bh);
      pb[0] = *reinterpret_cast<const uint4*>(bh_row + c * 32);
      pb[1] = *reinterpret_cast<const uint4*>(bh_row + c * 32 + 16);
      uint4* pl = reinterpret_cast<uint4*>(&bl);
      pl[0] = *reinterpret_cast<const uint4*>(bl_row + c * 32);
      pl[1] = *reinterpret_cast<const uint4*>(bl_row + c * 32 + 16);
      acc = __builtin_amdgcn_wmma_f32_16x16x32_bf16(false, a_hi[c], false, bh,
                                                    (short)0, acc, false, false);
      acc = __builtin_amdgcn_wmma_f32_16x16x32_bf16(false, a_hi[c], false, bl,
                                                    (short)0, acc, false, false);
      acc = __builtin_amdgcn_wmma_f32_16x16x32_bf16(false, a_lo[c], false, bh,
                                                    (short)0, acc, false, false);
    }

    // C/D layout: VGPR j holds (row = j or j+8 by lane half, col = lane&15).
    const float en  = enorm[nBase + bcol];
    const int   col = nBase + bcol;
#pragma unroll
    for (int j = 0; j < 8; ++j) {
      const float s = en - 2.0f * acc[j];
      if (s < minv[j]) { minv[j] = s; mini[j] = col; }   // strict < keeps first
    }
  }

  // Min-reduce across the 16 columns (lanes within each 16-lane half);
  // tie-break to the smaller index to match argmin's first-hit semantics.
#pragma unroll
  for (int m = 1; m < 16; m <<= 1) {
#pragma unroll
    for (int j = 0; j < 8; ++j) {
      const float ov = __shfl_xor(minv[j], m, 32);
      const int   oi = __shfl_xor(mini[j], m, 32);
      if (ov < minv[j] || (ov == minv[j] && oi < mini[j])) {
        minv[j] = ov; mini[j] = oi;
      }
    }
  }

  // Publish per-wave candidates (16 rows each) and merge the two codebook
  // halves in LDS. smv[wave][row]: wave = mslot + 4*half.
  __shared__ float smv[8][16];
  __shared__ int   smi[8][16];
  if (lane == 0) {
#pragma unroll
    for (int j = 0; j < 8; ++j) { smv[wave][j] = minv[j]; smi[wave][j] = mini[j]; }
  }
  if (lane == 16) {
#pragma unroll
    for (int j = 0; j < 8; ++j) { smv[wave][8 + j] = minv[j]; smi[wave][8 + j] = mini[j]; }
  }
  __syncthreads();

  const int t = threadIdx.x;
  if (t < 64) {                      // one thread per (M-tile, row)
    const int ms  = t >> 4;          // 0..3
    const int row = t & 15;
    const float v0 = smv[ms][row];
    const float v1 = smv[ms + 4][row];
    // half-0 indices are all smaller, so ties must pick half 0.
    const int best = (v1 < v0) ? smi[ms + 4][row] : smi[ms][row];
    best_idx[blockIdx.x * 64 + ms * 16 + row] = best;
  }
}

// ---------------------------------------------------------------------------
// Kernel 4: gather z_q = embedding[idx], write output (b,d,h,w) and per-row
// partial sums of (z_q - z)^2. grid = NROW blocks, 256 threads.
// ---------------------------------------------------------------------------
__global__ void gather_loss_kernel(const float* __restrict__ z,
                                   const float* __restrict__ emb,
                                   const int* __restrict__ best_idx,
                                   float* __restrict__ out,
                                   float* __restrict__ partials) {
  const int n  = blockIdx.x;
  const int d  = threadIdx.x;
  const int b  = n >> 10;
  const int hw = n & 1023;
  const int idx = best_idx[n];
  const float q  = emb[(size_t)idx * DIM + d];
  const size_t zi = ((size_t)(b * DIM + d)) * HW + hw;
  const float zv = z[zi];
  out[zi] = q;                      // straight-through output == z_q numerically
  const float diff = q - zv;

  __shared__ float sm[DIM];
  sm[d] = diff * diff;
  __syncthreads();
  for (int s = DIM / 2; s > 0; s >>= 1) {
    if (d < s) sm[d] += sm[d + s];
    __syncthreads();
  }
  if (d == 0) partials[n] = sm[0];
}

// Deterministic final reduction: q_loss = (1+BETA) * mean((z_q - z)^2)
__global__ void loss_reduce_kernel(const float* __restrict__ partials,
                                   float* __restrict__ loss_out) {
  const int t = threadIdx.x;        // 256 threads, 1 block
  float s = 0.f;
  for (int i = t; i < NROW; i += 256) s += partials[i];
  __shared__ float sm[256];
  sm[t] = s;
  __syncthreads();
  for (int k = 128; k > 0; k >>= 1) {
    if (t < k) sm[t] += sm[t + k];
    __syncthreads();
  }
  if (t == 0)
    loss_out[0] = sm[0] * (1.0f + BETA) / (float)((size_t)NROW * DIM);
}

// ---------------------------------------------------------------------------
extern "C" void kernel_launch(void* const* d_in, const int* in_sizes, int n_in,
                              void* d_out, int out_size, void* d_ws, size_t ws_size,
                              hipStream_t stream) {
  (void)in_sizes; (void)n_in; (void)out_size; (void)ws_size;
  const float* z   = (const float*)d_in[0];   // (16,256,32,32)
  const float* emb = (const float*)d_in[1];   // (8192,256)

  char* ws = (char*)d_ws;
  const size_t MB = 1024 * 1024;
  uint16_t* eh       = (uint16_t*)(ws);                    // 4 MB
  uint16_t* el       = (uint16_t*)(ws + 4  * MB);          // 4 MB
  uint16_t* zh       = (uint16_t*)(ws + 8  * MB);          // 8 MB
  uint16_t* zl       = (uint16_t*)(ws + 16 * MB);          // 8 MB
  float*    enorm    = (float*)   (ws + 24 * MB);          // 32 KB
  int*      best_idx = (int*)     (ws + 24 * MB + 64  * 1024); // 64 KB
  float*    partials = (float*)   (ws + 24 * MB + 192 * 1024); // 64 KB

  float* out  = (float*)d_out;                   // 16*256*32*32 floats
  float* loss = out + (size_t)NROW * DIM;        // final scalar

  prep_embedding_kernel<<<KCB, DIM, 0, stream>>>(emb, eh, el, enorm);
  prep_z_kernel<<<NROW, DIM, 0, stream>>>(z, zh, zl);
  // 256 blocks: 4 M-tiles per block x 2 codebook halves = 8 waves/block.
  vq_argmin_kernel<<<NROW / 64, 256, 0, stream>>>(zh, zl, eh, el, enorm,
                                                  best_idx);
  gather_loss_kernel<<<NROW, DIM, 0, stream>>>(z, emb, best_idx, out, partials);
  loss_reduce_kernel<<<1, 256, 0, stream>>>(partials, loss);
}
